// Decoder_45947560132795
// MI455X (gfx1250) — matrix operand order repaired and checked by reading, not compiled, not run on hardware
//
#include <hip/hip_runtime.h>
#include <hip/hip_bf16.h>

// ---------------------------------------------------------------------------
// Stacked tanh-RNN for MI455X (gfx1250, wave32, WMMA).
//   B=32, T=2048, NIN=256, H=512, L=4
// Phase A (per layer): xp = A @ W_ih^T + (b_ih + b_hh)   -- bf16 WMMA GEMM
// Phase B (per layer): h_t = tanh(xp_t + h_{t-1} @ W_hh^T)
//   4 persistent workgroups, 128 output cols each; h state double-buffered in
//   L2 (global bf16), step sync via agent-scope atomics; xp[t+1] staged into
//   LDS with global_load_async_to_lds_b128 (ASYNCcnt) while step t computes.
//   tanh via gfx1250 hardware TRANS op (v_tanh_f32) when the builtin exists.
// ---------------------------------------------------------------------------

typedef __bf16 bf16;
typedef __attribute__((ext_vector_type(16))) __bf16 v16bf;
typedef __attribute__((ext_vector_type(8)))  __bf16 v8bf;
typedef __attribute__((ext_vector_type(8)))  float  v8f;

static __device__ inline v16bf join16(v8bf lo, v8bf hi) {
  return __builtin_shufflevector(lo, hi, 0,1,2,3,4,5,6,7,8,9,10,11,12,13,14,15);
}

static __device__ inline float hw_tanh(float x) {
#if __has_builtin(__builtin_amdgcn_tanhf)
  return __builtin_amdgcn_tanhf(x);          // v_tanh_f32 (TRANS32, co-executes with WMMA)
#elif __has_builtin(__builtin_amdgcn_tanh_f32)
  return __builtin_amdgcn_tanh_f32(x);
#else
  return tanhf(x);
#endif
}

// ---------------------------------------------------------------------------
// f32 -> bf16 conversion
// ---------------------------------------------------------------------------
__global__ void __launch_bounds__(256) f32_to_bf16_kernel(const float* __restrict__ in,
                                                          bf16* __restrict__ out, int n) {
  int i = blockIdx.x * 256 + threadIdx.x;
  if (i < n) out[i] = (bf16)in[i];
}

// ---------------------------------------------------------------------------
// Input projection GEMM:  xp[t][b][n] = sum_k A[b*T+t][k] * W[n][k] + bias[n]
//   A: [M=65536, K] bf16 row-major,  W: [512, K] bf16 row-major
//   xp: time-major [T=2048][B=32][H=512] f32
// One wave computes a 16(M) x 64(N) tile: A-fragment reused across 4 WMMAs.
// ---------------------------------------------------------------------------
__global__ void __launch_bounds__(256) gemm_xp_kernel(const bf16* __restrict__ A,
                                                      const bf16* __restrict__ W,
                                                      const float* __restrict__ bih,
                                                      const float* __restrict__ bhh,
                                                      float* __restrict__ xp,
                                                      int K) {
  const int lane   = threadIdx.x & 31;
  const int wid    = blockIdx.x * 8 + (threadIdx.x >> 5);
  const int m_tile = wid >> 3;            // 0..4095 (16 rows each)
  const int n_quad = wid & 7;             // 0..7    (64 cols each)

  const int ncol  = lane & 15;
  const int aBase = (lane < 16) ? 0 : 8;  // A-frag K offset per half-wave
  const int bK    = (lane < 16) ? 0 : 8;  // B-frag K offset per half-wave
  const int n0q   = n_quad * 64;

  const int row = m_tile * 16 + ncol;
  const bf16* __restrict__ Arow = A + (size_t)row * K;

  v8f acc[4] = {};
  for (int k0 = 0; k0 < K; k0 += 32) {
    v8bf a0 = *(const v8bf*)(Arow + k0 + aBase);
    v8bf a1 = *(const v8bf*)(Arow + k0 + aBase + 16);
    v16bf afrag = join16(a0, a1);
#pragma unroll
    for (int j = 0; j < 4; ++j) {
      const bf16* Wp = W + (size_t)(n0q + j * 16 + ncol) * K + k0 + bK;
      v16bf bfrag = join16(*(const v8bf*)Wp, *(const v8bf*)(Wp + 16));
      acc[j] = __builtin_amdgcn_wmma_f32_16x16x32_bf16(
          false, afrag, false, bfrag, (short)0, acc[j], false, false);
    }
  }

  // C/D layout: VGPR v, lanes 0-15 -> M=v, N=lane; lanes 16-31 -> M=8+v, N=lane-16
  const int mAdd = (lane < 16) ? 0 : 8;
#pragma unroll
  for (int j = 0; j < 4; ++j) {
    const int n = n0q + j * 16 + ncol;
    const float bias = bih[n] + bhh[n];
#pragma unroll
    for (int v = 0; v < 8; ++v) {
      const int m = m_tile * 16 + v + mAdd;
      const int b = m >> 11;       // m / 2048
      const int t = m & 2047;      // m % 2048
      xp[((size_t)t * 32 + b) * 512 + n] = acc[j][v] + bias;
    }
  }
}

// ---------------------------------------------------------------------------
// Recurrent scan across 4 persistent workgroups.
//   WG g owns output columns [g*128, g*128+128).
//   Per step: 16 waves x 1 (16x16) tile x 16 K-steps = 256 WMMA per WG.
//   h state: hb0/hb1 global bf16 [32][512] double buffer (L2/L0-resident).
//   Step sync: agent-scope atomic counter; step t consumes hb[t&1] once
//   ctr >= 4*t (all WGs finished writing it at the end of step t-1).
//   xp[t+1] slice (32x128 f32 = 16KB) async-staged into LDS during step t.
// ---------------------------------------------------------------------------
__global__ void __launch_bounds__(512) rnn_scan4_kernel(const float* __restrict__ xp,   // [T][32][512]
                                                        const bf16* __restrict__ Whh,   // [512][512]
                                                        bf16* __restrict__ hb0,
                                                        bf16* __restrict__ hb1,
                                                        unsigned* __restrict__ ctr,
                                                        bf16* __restrict__ actOut,      // [B*T][512] or null
                                                        float* __restrict__ fOut) {     // [B][T][512] or null
  __shared__ float xbuf[2][32 * 128];              // 32 KB double buffer

  const int g    = blockIdx.x;                     // 0..3 column slice
  const int tid  = threadIdx.x;
  const int lane = tid & 31;
  const int w    = tid >> 5;                       // 0..15
  const int m0   = (w & 1) * 16;                   // M tile: rows 0..15 / 16..31
  const int nt   = w >> 1;                         // 0..7 local n-tile
  const int nloc0 = nt * 16;
  const int n0    = g * 128 + nloc0;

  const int ncol  = lane & 15;
  const int bK    = (lane < 16) ? 0 : 8;
  const int aBase = (lane < 16) ? 0 : 8;
  const int crow  = (lane < 16) ? 0 : 8;

  bf16* hbuf[2] = { hb0, hb1 };

  // Async-stage the 32x128 f32 slice of xp[t] into xbuf[t&1] via ASYNCcnt path.
  auto stage = [&](int t) {
    const float* src = xp + (size_t)t * (32 * 512) + g * 128;
#pragma unroll
    for (int i = 0; i < 2; ++i) {
      const int q  = tid + i * 512;                // 1024 chunks of 16B
      const int r  = q >> 5;                       // row 0..31
      const int cc = q & 31;                       // 16B chunk in row
      unsigned lds = (unsigned)(uintptr_t)&xbuf[t & 1][r * 128 + cc * 4];
      unsigned long long ga =
          (unsigned long long)(uintptr_t)(src + (size_t)r * 512 + cc * 4);
      asm volatile("global_load_async_to_lds_b128 %0, %1, off"
                   :: "v"(lds), "v"(ga) : "memory");
    }
  };

  stage(0);

  const bf16* __restrict__ Wbase = Whh + (size_t)(n0 + ncol) * 512 + bK;

  for (int t = 0; t < 2048; ++t) {
    const int buf = t & 1;

    if (t < 2047) {
      stage(t + 1);
      // async loads retire in order: <=2 outstanding => batch t has landed
      asm volatile("s_wait_asynccnt 0x2" ::: "memory");
    } else {
      asm volatile("s_wait_asynccnt 0x0" ::: "memory");
    }

    // wait for all 4 WGs to have produced h state for step t
    if (t > 0) {
      const unsigned target = 4u * (unsigned)t;
      while (__hip_atomic_load(ctr, __ATOMIC_ACQUIRE, __HIP_MEMORY_SCOPE_AGENT) < target)
        __builtin_amdgcn_s_sleep(1);
    }
    __syncthreads();                              // xbuf[buf] complete across waves

    // C init from staged xp
    v8f acc{};
#pragma unroll
    for (int v = 0; v < 8; ++v)
      acc[v] = xbuf[buf][(v + crow) * 128 + nloc0 + ncol];

    // acc += h @ Whh^T  (A-frags from L0-hot h buffer, B-frags L0-hot weights)
    const bf16* __restrict__ h = hbuf[buf];
#pragma unroll
    for (int kk = 0; kk < 16; ++kk) {
      const int k0 = kk * 32;
      const bf16* wp = Wbase + k0;
      v16bf bfrag = join16(*(const v8bf*)wp, *(const v8bf*)(wp + 16));
      const bf16* ap = h + (size_t)(m0 + ncol) * 512 + k0 + aBase;
      v16bf afrag = join16(*(const v8bf*)ap, *(const v8bf*)(ap + 16));
      acc = __builtin_amdgcn_wmma_f32_16x16x32_bf16(
          false, afrag, false, bfrag, (short)0, acc, false, false);
    }

    // tanh (hardware TRANS op) + publish h_{t+1} slice and layer outputs
    bf16* __restrict__ hn = hbuf[buf ^ 1];
#pragma unroll
    for (int v = 0; v < 8; ++v) {
      const float y = hw_tanh(acc[v]);
      const int m = m0 + v + crow;                 // batch row
      const int n = n0 + ncol;
      hn[m * 512 + n] = (bf16)y;
      if (actOut) actOut[((size_t)m * 2048 + t) * 512 + n] = (bf16)y;
      if (fOut)   fOut[((size_t)m * 2048 + t) * 512 + n] = y;
    }

    __threadfence();                               // agent-visible h writes
    __syncthreads();                               // whole WG done with step t
    if (tid == 0)
      __hip_atomic_fetch_add(ctr, 1u, __ATOMIC_RELEASE, __HIP_MEMORY_SCOPE_AGENT);
  }
}

// ---------------------------------------------------------------------------
// Host-side orchestration (all on `stream`, graph-capture safe)
// ---------------------------------------------------------------------------
extern "C" void kernel_launch(void* const* d_in, const int* in_sizes, int n_in,
                              void* d_out, int out_size, void* d_ws, size_t ws_size,
                              hipStream_t stream) {
  const float* x     = (const float*)d_in[0];  // [32,2048,256]
  const float* W_ih0 = (const float*)d_in[1];  // [512,256]
  const float* W_ih  = (const float*)d_in[2];  // [3,512,512]
  const float* W_hh  = (const float*)d_in[3];  // [4,512,512]
  const float* b_ih  = (const float*)d_in[4];  // [4,512]
  const float* b_hh  = (const float*)d_in[5];  // [4,512]
  float* out = (float*)d_out;                  // [32,2048,512]

  char* ws = (char*)d_ws;
  float* xp  = (float*)ws;  ws += (size_t)2048 * 32 * 512 * 4;   // 134.2 MB, [T][B][H]
  bf16* xb   = (bf16*)ws;   ws += (size_t)32 * 2048 * 256 * 2;   //  33.6 MB
  bf16* act  = (bf16*)ws;   ws += (size_t)32 * 2048 * 512 * 2;   //  67.1 MB
  bf16* wih0 = (bf16*)ws;   ws += (size_t)512 * 256 * 2;
  bf16* wih  = (bf16*)ws;   ws += (size_t)3 * 512 * 512 * 2;
  bf16* whh  = (bf16*)ws;   ws += (size_t)4 * 512 * 512 * 2;
  // scan sync state: hb0 | hb1 | ctr (zeroed per layer)
  char* hstate = ws;
  bf16* hb0 = (bf16*)hstate;                       // 32*512 bf16 = 32 KB
  bf16* hb1 = (bf16*)(hstate + 32768);             // 32 KB
  unsigned* ctr = (unsigned*)(hstate + 65536);     // + 256 B
  const size_t hstateBytes = 65536 + 256;

  auto cvt = [&](const float* src, bf16* dst, int n) {
    f32_to_bf16_kernel<<<(n + 255) / 256, 256, 0, stream>>>(src, dst, n);
  };
  cvt(x,     xb,   32 * 2048 * 256);
  cvt(W_ih0, wih0, 512 * 256);
  cvt(W_ih,  wih,  3 * 512 * 512);
  cvt(W_hh,  whh,  4 * 512 * 512);

  for (int l = 0; l < 4; ++l) {
    const bf16* A  = (l == 0) ? xb : act;
    const int   K  = (l == 0) ? 256 : 512;
    const bf16* Wl = (l == 0) ? wih0 : (wih + (size_t)(l - 1) * 512 * 512);
    gemm_xp_kernel<<<4096, 256, 0, stream>>>(A, Wl, b_ih + l * 512, b_hh + l * 512, xp, K);
    hipMemsetAsync(hstate, 0, hstateBytes, stream);   // h0 = 0, ctr = 0
    rnn_scan4_kernel<<<4, 512, 0, stream>>>(xp, whh + (size_t)l * 512 * 512,
                                            hb0, hb1, ctr,
                                            (l < 3) ? act : nullptr,
                                            (l == 3) ? out : nullptr);
  }
}
